// ESN_8667244003711
// MI455X (gfx1250) — matrix operand order repaired and checked by reading, not compiled, not run on hardware
//
#include <hip/hip_runtime.h>
#include <hip/hip_bf16.h>
#include <math.h>
#include <stdint.h>

// ---------------- problem constants ----------------
#define BB   8
#define TT   2000
#define DD   64
#define NN   2000
#define KP   2304   // padded K for recurrence: 2000 (state) + 64 (x) + pad = 72*32 = 8 waves * 9 chunks
#define NP2  2016   // padded K for output GEMM (h feature dim): 63*32
#define NWG  125    // 2000 / 16 column tiles
#define NWAVE 8
#define CPW  9      // K-chunks of 32 per wave

typedef __attribute__((ext_vector_type(16))) __bf16 v16bf;
typedef __attribute__((ext_vector_type(8)))  __bf16 v8bf;
typedef __attribute__((ext_vector_type(8)))  float  v8f;

// ---------------- workspace layout (bytes) ----------------
#define OFF_WT   ((size_t)0)                       // [2000][2304] bf16 = 9,216,000
#define OFF_WOT  (OFF_WT + (size_t)NN * KP * 2)    // [64][2016] bf16  =   258,048
#define OFF_A0   (OFF_WOT + (size_t)DD * NP2 * 2)  // [16][2304] bf16  =    73,728
#define OFF_A1   (OFF_A0 + (size_t)16 * KP * 2)
#define OFF_H    (OFF_A1 + (size_t)16 * KP * 2)    // [16000][2016] bf16 = 64,512,000
#define OFF_BAR  (OFF_H + (size_t)BB * TT * NP2 * 2)
#define ZERO_WORDS ((2 * (size_t)16 * KP * 2 + (size_t)BB * TT * NP2 * 2 + 256) / 4)

// ---------------- helpers ----------------
__global__ void k_zero(uint32_t* __restrict__ p, long n) {
    long i = (long)blockIdx.x * blockDim.x + threadIdx.x;
    long stride = (long)gridDim.x * blockDim.x;
    for (; i < n; i += stride) p[i] = 0u;
}

// WT[n][k] = k<2000 ? Wres[k][n] : k<2064 ? Win[k-2000][n] : 0   (bf16)
__global__ void k_fill_wt(const float* __restrict__ Wres,
                          const float* __restrict__ Win,
                          __bf16* __restrict__ WT) {
    int idx = blockIdx.x * 256 + threadIdx.x;
    if (idx >= NN * KP) return;
    int n = idx / KP;
    int k = idx - n * KP;
    float v = 0.0f;
    if (k < NN)            v = Wres[(size_t)k * NN + n];
    else if (k < NN + DD)  v = Win[(size_t)(k - NN) * NN + n];
    WT[idx] = (__bf16)v;
}

// WoutT[d][k] = k<2000 ? Wout[k][d] : 0   (bf16)
__global__ void k_fill_wot(const float* __restrict__ Wout,
                           __bf16* __restrict__ WoT) {
    int idx = blockIdx.x * 256 + threadIdx.x;
    if (idx >= DD * NP2) return;
    int d = idx / NP2;
    int k = idx - d * NP2;
    float v = (k < NN) ? Wout[(size_t)k * DD + d] : 0.0f;
    WoT[idx] = (__bf16)v;
}

// write x[:,0,:] into A0's x-columns
__global__ void k_x0(const float* __restrict__ x, __bf16* __restrict__ A0) {
    int tid = threadIdx.x;               // 512 threads
    int b = tid >> 6, j = tid & 63;
    A0[(size_t)b * KP + NN + j] = (__bf16)x[(size_t)b * TT * DD + j];
}

__device__ __forceinline__ void gridBarrier(unsigned* cnt, unsigned* gen) {
    __syncthreads();
    if (threadIdx.x == 0) {
        unsigned g = __hip_atomic_load(gen, __ATOMIC_RELAXED, __HIP_MEMORY_SCOPE_AGENT);
        unsigned arr = __hip_atomic_fetch_add(cnt, 1u, __ATOMIC_ACQ_REL, __HIP_MEMORY_SCOPE_AGENT);
        if (arr == NWG - 1) {
            __hip_atomic_store(cnt, 0u, __ATOMIC_RELAXED, __HIP_MEMORY_SCOPE_AGENT);
            __hip_atomic_fetch_add(gen, 1u, __ATOMIC_ACQ_REL, __HIP_MEMORY_SCOPE_AGENT);
        } else {
            while (__hip_atomic_load(gen, __ATOMIC_ACQUIRE, __HIP_MEMORY_SCOPE_AGENT) == g) {
                __builtin_amdgcn_s_sleep(2);
            }
        }
    }
    __syncthreads();
}

// branch-free tanh: 1 - 2/(exp(2x)+1); exact saturation at +-inf, monotone
__device__ __forceinline__ float fast_tanh(float v) {
    float e = __expf(2.0f * v);
    return 1.0f - 2.0f / (e + 1.0f);
}

// ---------------- persistent recurrence kernel ----------------
// grid = 125 WGs x 256 threads (8 waves). WG owns 16 reservoir columns.
// B (weight) fragments live in registers for the whole 2000-step run; per step
// each wave batch-issues all 18 A-fragment b128 loads (fenced so the scheduler
// cannot sink them into the WMMA chain), then chains 9 WMMAs whose waits
// retire the loads pipeline-style: one L2 latency per step, not nine.
__global__ void __launch_bounds__(256)
esn_recurrence(const float* __restrict__ x,
               const __bf16* __restrict__ WT,
               __bf16* __restrict__ A0,
               __bf16* __restrict__ A1,
               __bf16* __restrict__ h,
               unsigned* __restrict__ barCnt,
               unsigned* __restrict__ barGen) {
    const int tile = blockIdx.x;          // 0..124
    const int n0   = tile * 16;
    const int lane = threadIdx.x & 31;
    const int wave = threadIdx.x >> 5;

    __shared__ float red[(NWAVE - 1) * 256];   // 7 KB partial accumulators

    // fragment addressing (16-bit A 16x32 and B 32x16 VGPR layouts)
    const int aRow  = lane & 15;
    const int aKoff = (lane & 16) ? 8 : 0;
    const int bCol  = n0 + (lane & 15);
    const int bKoff = (lane & 16) ? 8 : 0;
    const int kBase = wave * CPW * 32;         // this wave's K origin

    // ---- load this wave's B fragments once; they never change across t ----
    v16bf Breg[CPW];
    {
        const __bf16* Bbase = WT + (size_t)bCol * KP + bKoff + kBase;
        #pragma unroll
        for (int c = 0; c < CPW; ++c) {
            v8bf lo = *(const v8bf*)(Bbase + c * 32);
            v8bf hi = *(const v8bf*)(Bbase + c * 32 + 16);
            Breg[c] = __builtin_shufflevector(lo, hi,
                      0,1,2,3,4,5,6,7,8,9,10,11,12,13,14,15);
        }
    }
    __builtin_amdgcn_sched_barrier(0);

    for (int t = 0; t < TT; ++t) {
        const __bf16* Ac = (t & 1) ? A1 : A0;
        __bf16*       An = (t & 1) ? A0 : A1;
        const __bf16* Arow = Ac + (size_t)aRow * KP + aKoff + kBase;

        // batch-issue every A load for this step (18 x global_load_b128)
        v8bf aLo[CPW], aHi[CPW];
        #pragma unroll
        for (int c = 0; c < CPW; ++c) {
            aLo[c] = *(const v8bf*)(Arow + c * 32);
            aHi[c] = *(const v8bf*)(Arow + c * 32 + 16);
        }
        // hard fence: no WMMA may be hoisted above, no load may sink below
        __builtin_amdgcn_sched_barrier(0);

        v8f acc = {0.f, 0.f, 0.f, 0.f, 0.f, 0.f, 0.f, 0.f};
        #pragma unroll
        for (int c = 0; c < CPW; ++c) {
            v16bf a = __builtin_shufflevector(aLo[c], aHi[c],
                      0,1,2,3,4,5,6,7,8,9,10,11,12,13,14,15);
            acc = __builtin_amdgcn_wmma_f32_16x16x32_bf16(
                false, a, false, Breg[c], (short)0, acc, false, false);
        }

        // cross-wave K reduction (waves 1..7 dump partials, wave 0 sums)
        if (wave != 0) {
            #pragma unroll
            for (int i = 0; i < 8; ++i) red[(wave - 1) * 256 + lane * 8 + i] = acc[i];
        }
        __syncthreads();
        if (wave == 0) {
            #pragma unroll
            for (int w = 0; w < NWAVE - 1; ++w)
                #pragma unroll
                for (int i = 0; i < 8; ++i) acc[i] += red[w * 256 + lane * 8 + i];
            if (lane < 16) {
                // C layout: VGPR i, lanes 0..15 -> M=i (real batch rows 0..7), N=lane
                const int col = n0 + lane;
                #pragma unroll
                for (int i = 0; i < 8; ++i) {
                    float s = fast_tanh(acc[i]);
                    An[(size_t)i * KP + col] = (__bf16)s;              // state for t+1
                    float g = ((col & 1) == 0) ? s * s : s;            // even-index square
                    h[((size_t)i * TT + t) * NP2 + col] = (__bf16)g;   // augmented feature
                }
            }
        }

        // WG0 stages x_{t+1} into the next A buffer (512 values / 256 threads)
        if (tile == 0 && (t + 1) < TT) {
            #pragma unroll
            for (int j = 0; j < 2; ++j) {
                int idx = threadIdx.x * 2 + j;
                int b_ = idx >> 6, d_ = idx & 63;
                An[(size_t)b_ * KP + NN + d_] =
                    (__bf16)x[((size_t)b_ * TT + (t + 1)) * DD + d_];
            }
        }

        __threadfence();
        gridBarrier(barCnt, barGen);
    }
}

// ---------------- output GEMM: out[16000,64] = h[16000,2016] @ WoutT^T ----------------
// grid = 1000 M-tiles * 4 N-tiles = 4000 single-wave WGs (occupancy hides latency)
__global__ void __launch_bounds__(32)
esn_out(const __bf16* __restrict__ h,
        const __bf16* __restrict__ WoT,
        float* __restrict__ out) {
    const int mt = blockIdx.x >> 2;
    const int nt = blockIdx.x & 3;
    const int lane = threadIdx.x & 31;

    const int aRow  = lane & 15;
    const int aKoff = (lane & 16) ? 8 : 0;
    const int bCol  = nt * 16 + (lane & 15);
    const int bKoff = (lane & 16) ? 8 : 0;
    const __bf16* Arow  = h + ((size_t)mt * 16 + aRow) * NP2 + aKoff;
    const __bf16* Bbase = WoT + (size_t)bCol * NP2 + bKoff;

    v8f acc = {0.f, 0.f, 0.f, 0.f, 0.f, 0.f, 0.f, 0.f};
    #pragma unroll 4
    for (int kc = 0; kc < 63; ++kc) {
        const int k0 = kc * 32;
        v8bf alo = *(const v8bf*)(Arow + k0);
        v8bf ahi = *(const v8bf*)(Arow + k0 + 16);
        v8bf blo = *(const v8bf*)(Bbase + k0);
        v8bf bhi = *(const v8bf*)(Bbase + k0 + 16);
        v16bf a = __builtin_shufflevector(alo, ahi, 0,1,2,3,4,5,6,7,8,9,10,11,12,13,14,15);
        v16bf b = __builtin_shufflevector(blo, bhi, 0,1,2,3,4,5,6,7,8,9,10,11,12,13,14,15);
        acc = __builtin_amdgcn_wmma_f32_16x16x32_bf16(
            false, a, false, b, (short)0, acc, false, false);
    }

    // C layout: VGPR i: lanes 0..15 -> M=i, lanes 16..31 -> M=8+i
    const int col   = nt * 16 + (lane & 15);
    const int rbase = (lane & 16) ? 8 : 0;
    #pragma unroll
    for (int i = 0; i < 8; ++i) {
        out[((size_t)mt * 16 + rbase + i) * DD + col] = acc[i];
    }
}

// ---------------- launch ----------------
extern "C" void kernel_launch(void* const* d_in, const int* in_sizes, int n_in,
                              void* d_out, int out_size, void* d_ws, size_t ws_size,
                              hipStream_t stream) {
    const float* x    = (const float*)d_in[0];
    const float* Win  = (const float*)d_in[1];
    const float* Wres = (const float*)d_in[2];
    const float* Wout = (const float*)d_in[3];
    float* out = (float*)d_out;

    char* ws = (char*)d_ws;
    __bf16* WT  = (__bf16*)(ws + OFF_WT);
    __bf16* WoT = (__bf16*)(ws + OFF_WOT);
    __bf16* A0  = (__bf16*)(ws + OFF_A0);
    __bf16* A1  = (__bf16*)(ws + OFF_A1);
    __bf16* H   = (__bf16*)(ws + OFF_H);
    unsigned* barCnt = (unsigned*)(ws + OFF_BAR);
    unsigned* barGen = (unsigned*)(ws + OFF_BAR + 64);

    // zero A buffers + h + barrier state (one grid-stride pass over contiguous region)
    k_zero<<<1024, 256, 0, stream>>>((uint32_t*)(ws + OFF_A0), (long)ZERO_WORDS);
    // build transposed bf16 weight blocks
    k_fill_wt<<<(NN * KP + 255) / 256, 256, 0, stream>>>(Wres, Win, WT);
    k_fill_wot<<<(DD * NP2 + 255) / 256, 256, 0, stream>>>(Wout, WoT);
    // stage x_0 into A0
    k_x0<<<1, 512, 0, stream>>>(x, A0);
    // persistent recurrence over 2000 steps with grid-wide barrier per step
    esn_recurrence<<<NWG, 256, 0, stream>>>(x, WT, A0, A1, H, barCnt, barGen);
    // final projection GEMM
    esn_out<<<4000, 32, 0, stream>>>(H, WoT, out);
}
